// PointNetFeaturePropagation_40140764348417
// MI455X (gfx1250) — matrix operand order repaired and checked by reading, not compiled, not run on hardware
//
#include <hip/hip_runtime.h>

// ---------------------------------------------------------------------------
// Types for CDNA5 WMMA (wave32)
// ---------------------------------------------------------------------------
typedef __attribute__((ext_vector_type(16))) __bf16 v16bf;
typedef __attribute__((ext_vector_type(8)))  __bf16 v8bf;
typedef __attribute__((ext_vector_type(8)))  float  v8f;
typedef __attribute__((ext_vector_type(4)))  int    v4i;

union BfPack { v16bf v; v8bf h[2]; };

// Problem constants
#define BATCH 4
#define NPTS  8192
#define SPTS  2048
#define D1C   256
#define D2C   512
#define KFEAT 768          // D1 + D2
#define MOUT  256
#define COLS  (BATCH * NPTS)   // 32768
#define BN_EPS_F 1e-5f

// ---------------------------------------------------------------------------
// Async global->LDS (CDNA5 GLOBAL_LOAD_ASYNC_TO_LDS_B128, ASYNCcnt-tracked).
// Builtin signature (from compiler diagnostic): first arg is v4i* (int4*).
// Guarded: falls back to a synchronous 16B copy if the builtin is absent.
// ---------------------------------------------------------------------------
#if defined(__has_builtin)
#if __has_builtin(__builtin_amdgcn_global_load_async_to_lds_b128)
#define HAVE_ASYNC_LDS 1
#endif
#endif
#ifndef HAVE_ASYNC_LDS
#define HAVE_ASYNC_LDS 0
#endif

typedef __attribute__((address_space(1))) v4i* gvec_t;
typedef __attribute__((address_space(3))) v4i* lvec_t;

__device__ __forceinline__ void async_copy16(const void* g, void* l) {
#if HAVE_ASYNC_LDS
  __builtin_amdgcn_global_load_async_to_lds_b128((gvec_t)g, (lvec_t)l, 0, 0);
#else
  *(uint4*)l = *(const uint4*)g;
#endif
}

#if HAVE_ASYNC_LDS
#define ASYNC_WAIT(n) asm volatile("s_wait_asynccnt %0" ::"n"(n) : "memory")
#else
#define ASYNC_WAIT(n)
#endif

// ---------------------------------------------------------------------------
// Kernel 1: convert fp32 weights -> bf16
// ---------------------------------------------------------------------------
__global__ __launch_bounds__(256) void cvt_weights(const float* __restrict__ W0,
                                                   const float* __restrict__ W1,
                                                   __bf16* __restrict__ Wb0,
                                                   __bf16* __restrict__ Wb1) {
  int i = blockIdx.x * 256 + threadIdx.x;
  if (i < MOUT * KFEAT) Wb0[i] = (__bf16)W0[i];
  if (i < MOUT * MOUT)  Wb1[i] = (__bf16)W1[i];
}

// Kernel: zero the BN stats scratch (4 x 256 floats)
__global__ __launch_bounds__(256) void zero_stats(float* __restrict__ s) {
  int t = threadIdx.x;
#pragma unroll
  for (int j = 0; j < 4; ++j) s[j * 256 + t] = 0.0f;
}

// ---------------------------------------------------------------------------
// Kernel 2: 3-NN interpolation + concat -> bf16 feature matrix (COLS x KFEAT)
// Point-major layout: feat[col*KFEAT + k], k contiguous (WMMA B-matrix order).
// ---------------------------------------------------------------------------
__global__ __launch_bounds__(256) void interp_concat(
    const float* __restrict__ xyz1,     // (B, N, 3)
    const float* __restrict__ xyz2,     // (B, S, 3)
    const float* __restrict__ points1,  // (B, D1, N)
    const float* __restrict__ points2,  // (B, D2, S)
    __bf16* __restrict__ feat)          // (COLS, KFEAT)
{
  __shared__ float sx[SPTS * 3];        // 24 KB of the 320 KB WGP LDS

  const int b     = blockIdx.x >> 5;          // 32 blocks per batch
  const int nbase = (blockIdx.x & 31) << 8;   // 256 points per block

  const float* x2 = xyz2 + (size_t)b * SPTS * 3;
  for (int i = threadIdx.x; i < SPTS * 3; i += 256) sx[i] = x2[i];
  __syncthreads();

  const int n   = nbase + threadIdx.x;
  const float* p = xyz1 + ((size_t)b * NPTS + n) * 3;
  const float px = p[0], py = p[1], pz = p[2];
  const float n1 = px * px + py * py + pz * pz;

  float d0 = 3.4e38f, d1 = 3.4e38f, d2 = 3.4e38f;
  int   i0 = 0, i1 = 0, i2 = 0;

  for (int s = 0; s < SPTS; ++s) {
    const float qx = sx[3 * s + 0];
    const float qy = sx[3 * s + 1];
    const float qz = sx[3 * s + 2];
    const float n2  = qx * qx + qy * qy + qz * qz;
    const float dot = px * qx + py * qy + pz * qz;
    const float d   = n1 + n2 - 2.0f * dot;       // same formula as reference
    if (d < d2) {
      if (d < d0)      { d2 = d1; i2 = i1; d1 = d0; i1 = i0; d0 = d; i0 = s; }
      else if (d < d1) { d2 = d1; i2 = i1; d1 = d;  i1 = s; }
      else             { d2 = d;  i2 = s; }
    }
  }

  const float r0 = 1.0f / (d0 + 1e-8f);
  const float r1 = 1.0f / (d1 + 1e-8f);
  const float r2 = 1.0f / (d2 + 1e-8f);
  const float rs = 1.0f / (r0 + r1 + r2);
  const float w0 = r0 * rs, w1 = r1 * rs, w2 = r2 * rs;

  const size_t col = (size_t)b * NPTS + n;
  __bf16* f = feat + col * KFEAT;

  // first D1 channels: transpose-copy of points1
  const float* p1c = points1 + (size_t)b * D1C * NPTS + n;
#pragma unroll 4
  for (int c = 0; c < D1C; ++c)
    f[c] = (__bf16)p1c[(size_t)c * NPTS];

  // next D2 channels: weighted gather from points2
  const float* p2 = points2 + (size_t)b * D2C * SPTS;
#pragma unroll 4
  for (int c = 0; c < D2C; ++c) {
    const float* row = p2 + (size_t)c * SPTS;
    const float v = w0 * row[i0] + w1 * row[i1] + w2 * row[i2];
    f[D1C + c] = (__bf16)v;
  }
}

// ---------------------------------------------------------------------------
// Kernel 3: bf16 GEMM via v_wmma_f32_16x16x32_bf16 with LDS-staged B tiles.
//   X : (cols, K) bf16, point-major (K contiguous)    -> WMMA B tiles (LDS)
//   W : (256, K)  bf16, row-major   (K contiguous)    -> WMMA A tiles (L2)
//   Y : (cols, 256) fp32                               <- WMMA D tiles
// One block = full M=256 x 64-column stripe. Each X tile is fetched from
// memory exactly once (async global->LDS, double buffered); 8 waves each
// compute 2 M-tiles x 4 N-tiles (8 WMMA per K-step).
// ---------------------------------------------------------------------------
#define NTILE          64
#define LDS_COL_STRIDE 80                       // 64B data + 16B pad (bank spread)
#define TILE_LDS_BYTES (NTILE * LDS_COL_STRIDE) // 5120

__global__ __launch_bounds__(256) void gemm_bf16_lds(
    const __bf16* __restrict__ X,
    const __bf16* __restrict__ W,
    float* __restrict__ Y,
    int K)
{
  __shared__ __align__(16) unsigned char smem[2 * TILE_LDS_BYTES];

  const int lane = threadIdx.x & 31;
  const int wave = threadIdx.x >> 5;          // 0..7 -> M rows 32w..32w+31
  const int half = lane >> 4;                 // 0: lanes 0-15, 1: lanes 16-31
  const int l15  = lane & 15;
  const int colBase = blockIdx.x * NTILE;

  // Staging: 256 threads x 16B = one 64col x 32K bf16 tile (4 KB)
  const int sCol   = threadIdx.x >> 2;        // 0..63
  const int sChunk = threadIdx.x & 3;         // 0..3 (16B chunks)
  const __bf16* gsrc  = X + (size_t)(colBase + sCol) * K + sChunk * 8;
  const int     ldOff = sCol * LDS_COL_STRIDE + sChunk * 16;

  // A-matrix layout (16-bit, 16x32): lane m holds row m.
  //   lanes 0-15 : K = k0..k0+7   and k0+16..k0+23
  //   lanes 16-31: K = k0+8..k0+15 and k0+24..k0+31
  const __bf16* Arow0 = W + (size_t)(32 * wave + l15) * K + (half ? 8 : 0);
  const __bf16* Arow1 = Arow0 + (size_t)16 * K;

  v8f acc[2][4] = {};

  const int nk = K >> 5;
  async_copy16(gsrc, smem + ldOff);           // prefetch tile 0

  for (int i = 0; i < nk; ++i) {
    const int cur = i & 1;
    if (i + 1 < nk) {
      async_copy16(gsrc + (size_t)(i + 1) * 32,
                   smem + ((i + 1) & 1) * TILE_LDS_BYTES + ldOff);
      ASYNC_WAIT(1);                          // tile i complete (this wave)
    } else {
      ASYNC_WAIT(0);
    }
    __syncthreads();                          // tile i complete (all waves)

    const int kb = i * 32;
    BfPack a0, a1;
    a0.h[0] = *(const v8bf*)(Arow0 + kb);
    a0.h[1] = *(const v8bf*)(Arow0 + kb + 16);
    a1.h[0] = *(const v8bf*)(Arow1 + kb);
    a1.h[1] = *(const v8bf*)(Arow1 + kb + 16);

    const unsigned char* bbuf = smem + cur * TILE_LDS_BYTES;
#pragma unroll
    for (int j = 0; j < 4; ++j) {
      // B layout (16-bit, 32x16): lane n holds column n;
      // lanes 0-15: K=k0..k0+7,k0+16..k0+23; lanes 16-31: K=k0+8..k0+15,k0+24..k0+31
      const unsigned char* bp =
          bbuf + (j * 16 + l15) * LDS_COL_STRIDE + half * 16;
      BfPack b;
      b.h[0] = *(const v8bf*)(bp);
      b.h[1] = *(const v8bf*)(bp + 32);
      acc[0][j] = __builtin_amdgcn_wmma_f32_16x16x32_bf16(
          false, a0.v, false, b.v, (short)0, acc[0][j], false, false);
      acc[1][j] = __builtin_amdgcn_wmma_f32_16x16x32_bf16(
          false, a1.v, false, b.v, (short)0, acc[1][j], false, false);
    }
    __syncthreads();                          // protect buffer reuse
  }

  // D layout (32-bit 16x16): VGPR r, lane l -> M = r + (l<16?0:8), N = l&15
#pragma unroll
  for (int tm = 0; tm < 2; ++tm) {
    const int mBase = 32 * wave + tm * 16 + (half ? 8 : 0);
#pragma unroll
    for (int j = 0; j < 4; ++j) {
      const size_t cb = (size_t)(colBase + j * 16 + l15) * MOUT;
#pragma unroll
      for (int r = 0; r < 8; ++r)
        Y[cb + mBase + r] = acc[tm][j][r];
    }
  }
}

// ---------------------------------------------------------------------------
// Kernel 4: per-channel sum / sum-of-squares over columns (for BatchNorm)
// Y is (cols, 256); thread c of each block strides over a chunk of columns.
// ---------------------------------------------------------------------------
__global__ __launch_bounds__(256) void channel_stats(
    const float* __restrict__ Y, float* __restrict__ sum,
    float* __restrict__ sumsq, int colsPerBlock)
{
  const int c = threadIdx.x;                // channel 0..255
  size_t base = (size_t)blockIdx.x * colsPerBlock * MOUT + c;
  float s = 0.0f, q = 0.0f;
  for (int i = 0; i < colsPerBlock; ++i) {
    const float v = Y[base + (size_t)i * MOUT];
    s += v;
    q += v * v;
  }
  atomicAdd(&sum[c], s);
  atomicAdd(&sumsq[c], q);
}

// ---------------------------------------------------------------------------
// Kernel 5: BN + ReLU -> bf16 activations for the next GEMM (same layout)
// ---------------------------------------------------------------------------
__global__ __launch_bounds__(256) void bn_relu_mid(
    const float* __restrict__ Y, const float* __restrict__ sum,
    const float* __restrict__ sumsq, const float* __restrict__ g,
    const float* __restrict__ bias, __bf16* __restrict__ Xn)
{
  const size_t i = (size_t)blockIdx.x * 256 + threadIdx.x;   // col*256 + c
  const int c = (int)(i & 255);
  const float invCount = 1.0f / (float)COLS;
  const float mean = sum[c] * invCount;
  const float var  = sumsq[c] * invCount - mean * mean;
  const float v = (Y[i] - mean) * rsqrtf(var + BN_EPS_F) * g[c] + bias[c];
  Xn[i] = (__bf16)fmaxf(v, 0.0f);
}

// ---------------------------------------------------------------------------
// Kernel 6: BN + ReLU -> final fp32 output in (B, 256, N) layout
// ---------------------------------------------------------------------------
__global__ __launch_bounds__(256) void bn_relu_out(
    const float* __restrict__ Y, const float* __restrict__ sum,
    const float* __restrict__ sumsq, const float* __restrict__ g,
    const float* __restrict__ bias, float* __restrict__ out)
{
  const size_t i = (size_t)blockIdx.x * 256 + threadIdx.x;   // col*256 + c
  const int    c   = (int)(i & 255);
  const size_t col = i >> 8;
  const float invCount = 1.0f / (float)COLS;
  const float mean = sum[c] * invCount;
  const float var  = sumsq[c] * invCount - mean * mean;
  const float v = (Y[i] - mean) * rsqrtf(var + BN_EPS_F) * g[c] + bias[c];
  const size_t b = col >> 13;          // /8192
  const size_t n = col & (NPTS - 1);
  out[((b << 8) + (size_t)c) * NPTS + n] = fmaxf(v, 0.0f);
}

// ---------------------------------------------------------------------------
// Host launcher
// ---------------------------------------------------------------------------
extern "C" void kernel_launch(void* const* d_in, const int* in_sizes, int n_in,
                              void* d_out, int out_size, void* d_ws, size_t ws_size,
                              hipStream_t stream) {
  (void)in_sizes; (void)n_in; (void)out_size; (void)ws_size;

  const float* xyz1    = (const float*)d_in[0];
  const float* xyz2    = (const float*)d_in[1];
  const float* points1 = (const float*)d_in[2];
  const float* points2 = (const float*)d_in[3];
  const float* W0      = (const float*)d_in[4];
  const float* g0      = (const float*)d_in[5];
  const float* b0      = (const float*)d_in[6];
  const float* W1      = (const float*)d_in[7];
  const float* g1      = (const float*)d_in[8];
  const float* b1      = (const float*)d_in[9];
  float* out = (float*)d_out;

  // Workspace carve-up (all offsets 256B aligned)
  char* ws = (char*)d_ws;
  size_t off = 0;
  __bf16* Wb0   = (__bf16*)(ws + off); off += (size_t)MOUT * KFEAT * 2;   // 384 KB
  __bf16* Wb1   = (__bf16*)(ws + off); off += (size_t)MOUT * MOUT  * 2;   // 128 KB
  float*  stats = (float*)(ws + off);  off += 4 * 256 * sizeof(float);    // sum0,sq0,sum1,sq1
  off = (off + 255) & ~(size_t)255;
  __bf16* Xfeat = (__bf16*)(ws + off); off += (size_t)COLS * KFEAT * 2;   // 48 MB
  float*  Y     = (float*) (ws + off); off += (size_t)COLS * MOUT * 4;    // 32 MB (reused by layer 1)
  __bf16* X1    = (__bf16*)(ws + off); off += (size_t)COLS * MOUT * 2;    // 16 MB

  float* sum0 = stats;       float* sq0 = stats + 256;
  float* sum1 = stats + 512; float* sq1 = stats + 768;

  // Pipeline
  cvt_weights<<<(MOUT * KFEAT + 255) / 256, 256, 0, stream>>>(W0, W1, Wb0, Wb1);
  zero_stats<<<1, 256, 0, stream>>>(stats);

  interp_concat<<<BATCH * 32, 256, 0, stream>>>(xyz1, xyz2, points1, points2, Xfeat);

  // Layer 0: (256 x 768) x (768 x 32768)
  gemm_bf16_lds<<<COLS / NTILE, 256, 0, stream>>>(Xfeat, Wb0, Y, KFEAT);
  channel_stats<<<COLS / 128, 256, 0, stream>>>(Y, sum0, sq0, 128);
  bn_relu_mid<<<COLS, 256, 0, stream>>>(Y, sum0, sq0, g0, b0, X1);

  // Layer 1: (256 x 256) x (256 x 32768)
  gemm_bf16_lds<<<COLS / NTILE, 256, 0, stream>>>(X1, Wb1, Y, MOUT);
  channel_stats<<<COLS / 128, 256, 0, stream>>>(Y, sum1, sq1, 128);
  bn_relu_out<<<COLS, 256, 0, stream>>>(Y, sum1, sq1, g1, b1, out);
}
